// combined_model_44367012168263
// MI455X (gfx1250) — matrix operand order repaired and checked by reading, not compiled, not run on hardware
//
#include <hip/hip_runtime.h>

// ---------------------------------------------------------------------------
// Types for CDNA5 WMMA / TDM
// ---------------------------------------------------------------------------
typedef __attribute__((ext_vector_type(16))) _Float16 v16h;
typedef __attribute__((ext_vector_type(8)))  float    v8f;
typedef __attribute__((ext_vector_type(4)))  unsigned int tdm_g0_t;
typedef __attribute__((ext_vector_type(8)))  int          tdm_g1_t;
typedef __attribute__((ext_vector_type(4)))  int          tdm_g23_t;
typedef __attribute__((ext_vector_type(8)))  int          tdm_g4_t;

__device__ __forceinline__ v8f wmma_f16(v16h a, v16h b, v8f c) {
  // D = A(16x32 f16) * B(32x16 f16) + C(16x16 f32)
  return __builtin_amdgcn_wmma_f32_16x16x32_f16(
      /*neg_a=*/false, a, /*neg_b=*/false, b,
      /*c_mod=*/(short)0, c, /*reuse_a=*/false, /*reuse_b=*/false);
}

// A-matrix fragment (16x32, f16): lanes 0-15 rows 0-15 hold K[0..7],K[16..23];
// lanes 16-31 hold K[8..15],K[24..31] for the same rows.
__device__ __forceinline__ v16h load_afrag(const _Float16* base, int ld, int row,
                                           int grp, int kc) {
  v16h a;
#pragma unroll
  for (int h = 0; h < 8; ++h) {
    a[h]     = base[row * ld + kc + grp * 8 + h];
    a[h + 8] = base[row * ld + kc + 16 + grp * 8 + h];
  }
  return a;
}

// B-matrix fragment (32x16, f16), B[k][n] = W[n][k] (weights row-major [N][K]):
// lanes 0-15 hold K[0..7],K[16..23]; lanes 16-31 hold K[8..15],K[24..31] for column n = lane&15.
__device__ __forceinline__ v16h load_bfrag(const _Float16* base, int ld, int n,
                                           int grp, int kc) {
  v16h b;
#pragma unroll
  for (int h = 0; h < 8; ++h) {
    b[h]     = base[n * ld + kc + grp * 8 + h];
    b[h + 8] = base[n * ld + kc + 16 + grp * 8 + h];
  }
  return b;
}

// 2D Tensor-Data-Mover load: tile (tile_d1 rows x tile_d0 elems) of a row-major
// tensor with row stride stride_elems into contiguous LDS at lds_off.
// data_size_code: 0=1B,1=2B,2=4B,3=8B.
__device__ __forceinline__ void tdm_load_2d(unsigned lds_off, const void* gaddr,
                                            unsigned tile_d0, unsigned tile_d1,
                                            unsigned tensor_d0, unsigned tensor_d1,
                                            unsigned stride_elems,
                                            unsigned data_size_code) {
  unsigned long long ga = (unsigned long long)gaddr;
  tdm_g0_t g0;
  g0[0] = 1u;                                   // count=1, user descriptor
  g0[1] = lds_off;                              // LDS byte address
  g0[2] = (unsigned)(ga & 0xFFFFFFFFu);         // global addr [31:0]
  g0[3] = (unsigned)((ga >> 32) & 0x01FFFFFFu)  // global addr [56:32]
          | 0x80000000u;                        // type = 2 ("image")
  tdm_g1_t g1;
  g1[0] = (int)(data_size_code << 16);          // wg_mask=0, data_size
  g1[1] = (int)((tensor_d0 & 0xFFFFu) << 16);   // abar=0 | tensor_dim0 lo16
  g1[2] = (int)((tensor_d0 >> 16) | ((tensor_d1 & 0xFFFFu) << 16));
  g1[3] = (int)((tensor_d1 >> 16) | (tile_d0 << 16));
  g1[4] = (int)(tile_d1 & 0xFFFFu);             // tile_dim1 | tile_dim2=0
  g1[5] = (int)stride_elems;                    // tensor_dim0_stride lo32
  g1[6] = 0;
  g1[7] = 0;
  tdm_g23_t z  = {0, 0, 0, 0};
  tdm_g4_t  z8 = {0, 0, 0, 0, 0, 0, 0, 0};
  __builtin_amdgcn_tensor_load_to_lds(g0, g1, z, z, z8, 0);
}

__device__ __forceinline__ float sigmf(float x) { return 1.f / (1.f + __expf(-x)); }

// ---------------------------------------------------------------------------
// Model dims
// ---------------------------------------------------------------------------
#define BB   128
#define TT   8
#define LLAY 8
#define HID  64
// conv1: (3,192,256) -> (32,94,126), conv2 -> (64,46,62), conv3 -> (64,22,30)
#define C1OUT (128 * 32 * 94 * 126)
#define C2OUT (128 * 64 * 46 * 62)
#define C3K   42240
#define C3OUT (128 * C3K)

// ---------------------------------------------------------------------------
// K1: conv1 (Cin=3, 5x5, stride 2) + ReLU, fp32 VALU, f16 output (mem-bound)
// ---------------------------------------------------------------------------
__global__ void conv1_kernel(const float* __restrict__ in,
                             const float* __restrict__ w,
                             const float* __restrict__ b,
                             _Float16* __restrict__ out) {
  __shared__ float sw[32 * 75];
  __shared__ float sb[32];
  for (int i = threadIdx.x; i < 32 * 75; i += 256) sw[i] = w[i];
  if (threadIdx.x < 32) sb[threadIdx.x] = b[threadIdx.x];
  __syncthreads();
  int idx = blockIdx.x * 256 + threadIdx.x;
  int ow = idx % 126; int t = idx / 126;
  int oh = t % 94;  t /= 94;
  int oc = t % 32;  int bi = t / 32;
  const float* ip = in + ((size_t)(bi * 3) * 192 + oh * 2) * 256 + ow * 2;
  const float* wp = sw + oc * 75;
  float acc = sb[oc];
#pragma unroll
  for (int ic = 0; ic < 3; ++ic)
#pragma unroll
    for (int r = 0; r < 5; ++r)
#pragma unroll
      for (int s = 0; s < 5; ++s)
        acc = fmaf(ip[((size_t)ic * 192 + r) * 256 + s], wp[(ic * 5 + r) * 5 + s], acc);
  out[idx] = (_Float16)fmaxf(acc, 0.f);
}

// ---------------------------------------------------------------------------
// K2: implicit-GEMM conv (3x3, stride 2, Cout=64) via WMMA f16.
// Compile-time dims (no integer-division codegen). Full filter staged ONCE per
// block in dynamic LDS; block grid-strides over 16-pixel M-tiles; 4 waves =
// 4 x 16 output channels.
// ---------------------------------------------------------------------------
template <int CIN, int IH, int IW, int OH, int OW, int RELU>
__global__ void conv_wmma_kernel(const _Float16* __restrict__ in,
                                 const float* __restrict__ w,
                                 const float* __restrict__ bias,
                                 _Float16* __restrict__ out, int ntiles) {
  extern __shared__ _Float16 smem[];
  _Float16* sW = smem;                 // [9][64][CIN] all taps
  _Float16* sA = smem + 9 * 64 * CIN;  // [16][CIN]
  const int tid = threadIdx.x, lane = tid & 31, wv = tid >> 5;
  const int row = lane & 15, grp = lane >> 4;
  // Stage the whole filter once (L2-resident source).
  for (int i = tid; i < 9 * 64 * CIN; i += 128) {
    int tap = i / (64 * CIN), rem = i - tap * (64 * CIN);
    int oc = rem / CIN, ic = rem - oc * CIN;
    int r = tap / 3, s = tap - r * 3;
    sW[i] = (_Float16)w[((oc * CIN + ic) * 3 + r) * 3 + s];
  }
  const int oc = wv * 16 + (lane & 15);
  const float bv = bias[oc];
  __syncthreads();

  for (int mt = blockIdx.x; mt < ntiles; mt += gridDim.x) {
    const int pix0 = mt * 16;
    v8f acc = {};
#pragma unroll
    for (int tap = 0; tap < 9; ++tap) {
      const int r = tap / 3, s = tap - r * 3;
      __syncthreads();  // previous tap's wmma reads of sA complete
      for (int i = tid; i < 16 * CIN; i += 128) {
        int p = i / CIN, ic = i - p * CIN;
        int pix = pix0 + p;
        int owp = pix % OW; int t2 = pix / OW;
        int ohp = t2 % OH;  int bi = t2 / OH;
        sA[p * CIN + ic] =
            in[((size_t)(bi * CIN + ic) * IH + ohp * 2 + r) * IW + owp * 2 + s];
      }
      __syncthreads();
#pragma unroll
      for (int kc = 0; kc < CIN; kc += 32) {
        v16h a  = load_afrag(sA, CIN, row, grp, kc);
        v16h bf = load_bfrag(sW + tap * 64 * CIN, CIN, oc, grp, kc);
        acc = wmma_f16(a, bf, acc);
      }
    }
#pragma unroll
    for (int vr = 0; vr < 8; ++vr) {
      int pix = pix0 + vr + grp * 8;
      int owp = pix % OW; int t2 = pix / OW;
      int ohp = t2 % OH;  int bi = t2 / OH;
      float v = acc[vr] + bv;
      if (RELU) v = fmaxf(v, 0.f);
      out[((size_t)(bi * 64 + oc) * OH + ohp) * OW + owp] = (_Float16)v;
    }
  }
}

// ---------------------------------------------------------------------------
// K3: fp32 -> fp16 conversion (fc1 weights, done once; stays L2-resident)
// ---------------------------------------------------------------------------
__global__ void cvt_f32_to_f16_kernel(const float* __restrict__ in,
                                      _Float16* __restrict__ out, int n) {
  int i = blockIdx.x * blockDim.x + threadIdx.x;
  if (i < n) out[i] = (_Float16)in[i];
}

// ---------------------------------------------------------------------------
// K4: fc1 GEMM 128x42240x256 + ReLU. A (f16 conv3 out) and W (f16) K=64 chunks
// are DMA'd into LDS by the Tensor Data Mover, double-buffered so DMA of chunk
// i+1 overlaps the 2 WMMAs of chunk i. 16 waves = 256 output columns.
// Dynamic LDS: 2x(256x64 + 16x64) f16 = 68KB. (Scalar buffer pointers only:
// local pointer ARRAYS into extern-shared break ld.lld static initializers.)
// ---------------------------------------------------------------------------
__global__ void fc1_kernel(const _Float16* __restrict__ A,
                           const _Float16* __restrict__ Wh,
                           const float* __restrict__ bias,
                           float* __restrict__ out) {
  extern __shared__ _Float16 fsm[];
  _Float16* const sW0 = fsm;
  _Float16* const sW1 = fsm + 256 * 64;
  _Float16* const sA0 = fsm + 2 * 256 * 64;
  _Float16* const sA1 = fsm + 2 * 256 * 64 + 16 * 64;
  const int tid = threadIdx.x, lane = tid & 31, wv = tid >> 5;  // 512 thr, 16 waves
  const int row = lane & 15, grp = lane >> 4;
  const int m0 = blockIdx.x * 16;
  const unsigned ldsW0 = (unsigned)(unsigned long long)(void*)sW0;
  const unsigned ldsW1 = (unsigned)(unsigned long long)(void*)sW1;
  const unsigned ldsA0 = (unsigned)(unsigned long long)(void*)sA0;
  const unsigned ldsA1 = (unsigned)(unsigned long long)(void*)sA1;
  const int NCHUNK = C3K / 64;  // 660
  v8f acc = {};
  if (wv == 0) {  // prologue: chunk 0 into buffer 0
    tdm_load_2d(ldsA0, A + (size_t)m0 * C3K, 64, 16, C3K, BB, C3K, 1);
    tdm_load_2d(ldsW0, Wh, 64, 256, C3K, 256, C3K, 1);
  }
  for (int i = 0; i < NCHUNK; ++i) {
    const int cur = i & 1;
    if (wv == 0) {
      if (i + 1 < NCHUNK) {  // issue chunk i+1 into the other buffer
        const int kn = (i + 1) * 64;
        tdm_load_2d(cur ? ldsA0 : ldsA1, A + (size_t)m0 * C3K + kn,
                    64, 16, C3K, BB, C3K, 1);
        tdm_load_2d(cur ? ldsW0 : ldsW1, Wh + kn, 64, 256, C3K, 256, C3K, 1);
        __builtin_amdgcn_s_wait_tensorcnt(2);  // chunk i's 2 DMAs done
      } else {
        __builtin_amdgcn_s_wait_tensorcnt(0);
      }
    }
    __syncthreads();
    const _Float16* sWc = cur ? sW1 : sW0;
    const _Float16* sAc = cur ? sA1 : sA0;
    const int n = wv * 16 + (lane & 15);
#pragma unroll
    for (int kc = 0; kc < 64; kc += 32) {
      v16h a  = load_afrag(sAc, 64, row, grp, kc);
      v16h bf = load_bfrag(sWc, 64, n, grp, kc);
      acc = wmma_f16(a, bf, acc);
    }
    __syncthreads();  // all waves done with buffer `cur` before it is refilled
  }
  const int n = wv * 16 + (lane & 15);
  const float bv = bias[n];
#pragma unroll
  for (int vr = 0; vr < 8; ++vr) {
    int m = m0 + vr + grp * 8;
    out[(size_t)m * 256 + n] = fmaxf(acc[vr] + bv, 0.f);
  }
}

// ---------------------------------------------------------------------------
// K5: fc2 -> lowd1(+relu) -> lowd2 chain, WMMA, activations resident in LDS.
// 1 block, 8 waves; wave w owns 16-row M-tile, loops all 8 N-tiles.
// ---------------------------------------------------------------------------
__global__ void head_mlp_kernel(const float* __restrict__ x,    // 128x256 fc1 out
                                const float* __restrict__ fc2w, const float* __restrict__ fc2b,
                                const float* __restrict__ l1w,  const float* __restrict__ l1b,
                                const float* __restrict__ l2w,  const float* __restrict__ l2b,
                                float* __restrict__ cnn2) {     // 128x128
  __shared__ _Float16 sX[128 * 32];   // 8KB  A chunk (GEMM1 only)
  __shared__ _Float16 sW[128 * 32];   // 8KB  W chunk
  __shared__ _Float16 sO[128 * 128];  // 32KB activations
  const int tid = threadIdx.x, lane = tid & 31, wv = tid >> 5;  // 256 thr, 8 waves
  const int row = lane & 15, grp = lane >> 4;
  const int m0 = wv * 16;
  v8f acc[8] = {};

  // GEMM1: cnn_out1 = x @ fc2_w^T + b   (K=256, no relu)
  for (int k0 = 0; k0 < 256; k0 += 32) {
    __syncthreads();
    for (int i = tid; i < 128 * 32; i += 256) {
      int rr = i >> 5, kk = i & 31;
      sX[i] = (_Float16)x[rr * 256 + k0 + kk];
      sW[i] = (_Float16)fc2w[rr * 256 + k0 + kk];
    }
    __syncthreads();
    v16h a = load_afrag(sX + m0 * 32, 32, row, grp, 0);
#pragma unroll
    for (int nt = 0; nt < 8; ++nt) {
      v16h bf = load_bfrag(sW, 32, nt * 16 + (lane & 15), grp, 0);
      acc[nt] = wmma_f16(a, bf, acc[nt]);
    }
  }
  __syncthreads();
#pragma unroll
  for (int nt = 0; nt < 8; ++nt) {
    int n = nt * 16 + (lane & 15);
    float bv = fc2b[n];
#pragma unroll
    for (int vr = 0; vr < 8; ++vr) {
      sO[(m0 + vr + grp * 8) * 128 + n] = (_Float16)(acc[nt][vr] + bv);
      acc[nt][vr] = 0.f;
    }
  }
  __syncthreads();

  // GEMM2: relu(sO @ lowd1_w^T + b)  (K=128)
  for (int k0 = 0; k0 < 128; k0 += 32) {
    __syncthreads();
    for (int i = tid; i < 128 * 32; i += 256) {
      int rr = i >> 5, kk = i & 31;
      sW[i] = (_Float16)l1w[rr * 128 + k0 + kk];
    }
    __syncthreads();
    v16h a = load_afrag(sO + m0 * 128, 128, row, grp, k0);
#pragma unroll
    for (int nt = 0; nt < 8; ++nt) {
      v16h bf = load_bfrag(sW, 32, nt * 16 + (lane & 15), grp, 0);
      acc[nt] = wmma_f16(a, bf, acc[nt]);
    }
  }
  __syncthreads();
#pragma unroll
  for (int nt = 0; nt < 8; ++nt) {
    int n = nt * 16 + (lane & 15);
    float bv = l1b[n];
#pragma unroll
    for (int vr = 0; vr < 8; ++vr) {
      sO[(m0 + vr + grp * 8) * 128 + n] = (_Float16)fmaxf(acc[nt][vr] + bv, 0.f);
      acc[nt][vr] = 0.f;
    }
  }
  __syncthreads();

  // GEMM3: cnn_out2 = sO @ lowd2_w^T + b -> global f32
  for (int k0 = 0; k0 < 128; k0 += 32) {
    __syncthreads();
    for (int i = tid; i < 128 * 32; i += 256) {
      int rr = i >> 5, kk = i & 31;
      sW[i] = (_Float16)l2w[rr * 128 + k0 + kk];
    }
    __syncthreads();
    v16h a = load_afrag(sO + m0 * 128, 128, row, grp, k0);
#pragma unroll
    for (int nt = 0; nt < 8; ++nt) {
      v16h bf = load_bfrag(sW, 32, nt * 16 + (lane & 15), grp, 0);
      acc[nt] = wmma_f16(a, bf, acc[nt]);
    }
  }
#pragma unroll
  for (int nt = 0; nt < 8; ++nt) {
    int n = nt * 16 + (lane & 15);
    float bv = l2b[n];
#pragma unroll
    for (int vr = 0; vr < 8; ++vr)
      cnn2[(m0 + vr + grp * 8) * 128 + n] = acc[nt][vr] + bv;
  }
}

// ---------------------------------------------------------------------------
// K6: action MLP (2 -> 16 relu -> 16), written K-padded to 64 into xbuf0 f16
// ---------------------------------------------------------------------------
__global__ void action_kernel(const float* __restrict__ ain,
                              const float* __restrict__ w1, const float* __restrict__ b1,
                              const float* __restrict__ w2, const float* __restrict__ b2,
                              _Float16* __restrict__ xbuf0) {
  int idx = blockIdx.x * 256 + threadIdx.x;   // (b*8+t)*64 + h
  int h = idx & 63;
  int bt = idx >> 6;
  float v = 0.f;
  if (h < 16) {
    float x0 = ain[bt * 2 + 0], x1 = ain[bt * 2 + 1];
    float s = b2[h];
#pragma unroll
    for (int j = 0; j < 16; ++j) {
      float a1 = fmaxf(fmaf(w1[j * 2], x0, fmaf(w1[j * 2 + 1], x1, b1[j])), 0.f);
      s = fmaf(a1, w2[h * 16 + j], s);
    }
    v = s;
  }
  xbuf0[idx] = (_Float16)v;
}

// ---------------------------------------------------------------------------
// K7: 8-layer LSTM. 8 blocks x 16-batch slices; 4 waves/block, each wave owns
// a 16x16 (batch x hid) slice and all 4 gates; cell state c stays in VGPRs.
// Wih+Whh staged once per layer (66KB dynamic LDS); biases hoisted per layer.
// ---------------------------------------------------------------------------
__global__ void lstm_kernel(const float* __restrict__ cnn2,   // 128x128 (c | h)
                            const float* __restrict__ Wih0,   // 256x16
                            const float* __restrict__ Wih,    // 7x256x64
                            const float* __restrict__ Whh,    // 8x256x64
                            const float* __restrict__ bih,    // 8x256
                            const float* __restrict__ bhh,    // 8x256
                            _Float16* __restrict__ xbuf0,
                            _Float16* __restrict__ xbuf1) {
  extern __shared__ _Float16 lsm[];
  _Float16* sWih = lsm;                 // 256x64 (32KB)
  _Float16* sWhh = lsm + 256 * 64;      // 256x64 (32KB)
  _Float16* sH   = lsm + 2 * 256 * 64;  // 16x64  (2KB)
  const int tid = threadIdx.x, lane = tid & 31, wv = tid >> 5;  // 128 thr, 4 waves
  const int row = lane & 15, grp = lane >> 4;
  const int b0 = blockIdx.x * 16;
  const int hs = wv;                 // hid slice 0..3
  const int n16 = lane & 15;
  const int col = hs * 16 + n16;

  for (int l = 0; l < LLAY; ++l) {
    _Float16* xin  = (l & 1) ? xbuf1 : xbuf0;
    _Float16* xout = (l & 1) ? xbuf0 : xbuf1;
    const float* WihL = (l == 0) ? Wih0 : (Wih + (size_t)(l - 1) * 256 * 64);
    const int Kin = (l == 0) ? 16 : 64;
    __syncthreads();  // previous layer fully done with LDS
    // Stage layer weights ONCE (padded Wih for l==0) + h0; init c in VGPRs.
    for (int i = tid; i < 256 * 64; i += 128) {
      int n = i >> 6, k = i & 63;
      sWih[i] = (k < Kin) ? (_Float16)WihL[n * Kin + k] : (_Float16)0.f;
      sWhh[i] = (_Float16)Whh[(size_t)l * 256 * 64 + i];
    }
    for (int i = tid; i < 16 * 64; i += 128) {
      int rr = i >> 6, hh = i & 63;
      sH[i] = (_Float16)cnn2[(b0 + rr) * 128 + HID + hh];
    }
    v8f c;
#pragma unroll
    for (int vr = 0; vr < 8; ++vr)
      c[vr] = cnn2[(b0 + vr + grp * 8) * 128 + col];
    // Gate biases: constant over t for this (layer, col).
    const float bi = bih[l * 256 + col]       + bhh[l * 256 + col];
    const float bf = bih[l * 256 + 64 + col]  + bhh[l * 256 + 64 + col];
    const float bg = bih[l * 256 + 128 + col] + bhh[l * 256 + 128 + col];
    const float bo = bih[l * 256 + 192 + col] + bhh[l * 256 + 192 + col];
    __syncthreads();

    for (int t = 0; t < TT; ++t) {
      v8f g[4] = {};
      // gates = x_t @ Wih^T  (A read straight from global f16 sequence buffer)
      const _Float16* xrow = xin + ((size_t)(b0 + row) * TT + t) * 64;
#pragma unroll
      for (int kc = 0; kc < 64; kc += 32) {
        v16h a;
#pragma unroll
        for (int h = 0; h < 8; ++h) {
          a[h]     = xrow[kc + grp * 8 + h];
          a[h + 8] = xrow[kc + 16 + grp * 8 + h];
        }
#pragma unroll
        for (int gi = 0; gi < 4; ++gi) {
          v16h bfr = load_bfrag(sWih, 64, gi * 64 + col, grp, kc);
          g[gi] = wmma_f16(a, bfr, g[gi]);
        }
      }
      // gates += h_{t-1} @ Whh^T
#pragma unroll
      for (int kc = 0; kc < 64; kc += 32) {
        v16h a = load_afrag(sH, 64, row, grp, kc);
#pragma unroll
        for (int gi = 0; gi < 4; ++gi) {
          v16h bfr = load_bfrag(sWhh, 64, gi * 64 + col, grp, kc);
          g[gi] = wmma_f16(a, bfr, g[gi]);
        }
      }
      __syncthreads();  // everyone done reading sH
      // activations + state update (c stays in VGPRs)
#pragma unroll
      for (int vr = 0; vr < 8; ++vr) {
        float iv = sigmf(g[0][vr] + bi);
        float fv = sigmf(g[1][vr] + bf);
        float gv = tanhf(g[2][vr] + bg);
        float ov = sigmf(g[3][vr] + bo);
        c[vr] = fv * c[vr] + iv * gv;
        float hv = ov * tanhf(c[vr]);
        int m = vr + grp * 8;
        _Float16 h16 = (_Float16)hv;
        sH[m * 64 + col] = h16;
        xout[((size_t)(b0 + m) * TT + t) * 64 + col] = h16;
      }
      __syncthreads();
    }
  }
}

// ---------------------------------------------------------------------------
// K8: out1(relu) -> out2 -> yaw rotation + translation -> d_out (B,T,4)
// ---------------------------------------------------------------------------
__global__ void output_kernel(const _Float16* __restrict__ x,  // [b][t][64] f16
                              const float* __restrict__ w1, const float* __restrict__ b1,
                              const float* __restrict__ w2, const float* __restrict__ b2,
                              const float* __restrict__ gt,    // (128,8,6)
                              float* __restrict__ out) {
  __shared__ float sw1[32 * 64], sb1[32], sw2[4 * 32], sb2[4];
  for (int i = threadIdx.x; i < 32 * 64; i += 256) sw1[i] = w1[i];
  if (threadIdx.x < 32) sb1[threadIdx.x] = b1[threadIdx.x];
  if (threadIdx.x < 128) sw2[threadIdx.x] = w2[threadIdx.x];
  if (threadIdx.x < 4) sb2[threadIdx.x] = b2[threadIdx.x];
  __syncthreads();
  int idx = blockIdx.x * 256 + threadIdx.x;  // b*8 + t
  int b = idx >> 3;
  float h[64];
  const _Float16* xr = x + (size_t)idx * 64;
#pragma unroll
  for (int k = 0; k < 64; ++k) h[k] = (float)xr[k];
  float mo[4] = {sb2[0], sb2[1], sb2[2], sb2[3]};
  for (int j = 0; j < 32; ++j) {
    float a = sb1[j];
#pragma unroll
    for (int k = 0; k < 64; ++k) a = fmaf(h[k], sw1[j * 64 + k], a);
    a = fmaxf(a, 0.f);
#pragma unroll
    for (int o = 0; o < 4; ++o) mo[o] = fmaf(a, sw2[o * 32 + j], mo[o]);
  }
  float yaw = gt[b * 48 + 5];
  float cy = __cosf(yaw), sy = __sinf(yaw);
  out[idx * 4 + 0] = fmaf(mo[0], cy, mo[1] * sy) + gt[b * 48 + 1];
  out[idx * 4 + 1] = fmaf(-mo[0], sy, mo[1] * cy) + gt[b * 48 + 2];
  out[idx * 4 + 2] = mo[2] + gt[b * 48 + 3];
  out[idx * 4 + 3] = mo[3];
}

// ---------------------------------------------------------------------------
// Launch
// ---------------------------------------------------------------------------
extern "C" void kernel_launch(void* const* d_in, const int* in_sizes, int n_in,
                              void* d_out, int out_size, void* d_ws, size_t ws_size,
                              hipStream_t stream) {
  (void)in_sizes; (void)n_in; (void)out_size; (void)ws_size;
  const float* image   = (const float*)d_in[0];
  const float* act_in  = (const float*)d_in[1];
  const float* gt      = (const float*)d_in[2];
  const float* conv1_w = (const float*)d_in[3];
  const float* conv1_b = (const float*)d_in[4];
  const float* conv2_w = (const float*)d_in[5];
  const float* conv2_b = (const float*)d_in[6];
  const float* conv3_w = (const float*)d_in[7];
  const float* conv3_b = (const float*)d_in[8];
  const float* fc1_w   = (const float*)d_in[9];
  const float* fc1_b   = (const float*)d_in[10];
  const float* fc2_w   = (const float*)d_in[11];
  const float* fc2_b   = (const float*)d_in[12];
  const float* l1_w    = (const float*)d_in[13];
  const float* l1_b    = (const float*)d_in[14];
  const float* l2_w    = (const float*)d_in[15];
  const float* l2_b    = (const float*)d_in[16];
  const float* a1_w    = (const float*)d_in[17];
  const float* a1_b    = (const float*)d_in[18];
  const float* a2_w    = (const float*)d_in[19];
  const float* a2_b    = (const float*)d_in[20];
  const float* Wih0    = (const float*)d_in[21];
  const float* Wih     = (const float*)d_in[22];
  const float* Whh     = (const float*)d_in[23];
  const float* bih     = (const float*)d_in[24];
  const float* bhh     = (const float*)d_in[25];
  const float* o1_w    = (const float*)d_in[26];
  const float* o1_b    = (const float*)d_in[27];
  const float* o2_w    = (const float*)d_in[28];
  const float* o2_b    = (const float*)d_in[29];

  char* ws = (char*)d_ws;
  size_t off = 0;
  auto take = [&](size_t bytes) -> char* {
    char* p = ws + off;
    off += (bytes + 255) & ~(size_t)255;
    return p;
  };
  _Float16* c1o    = (_Float16*)take((size_t)C1OUT * 2);
  _Float16* c2o    = (_Float16*)take((size_t)C2OUT * 2);
  _Float16* c3o    = (_Float16*)take((size_t)C3OUT * 2);
  _Float16* fc1w16 = (_Float16*)take((size_t)256 * C3K * 2);
  float*    fc1o   = (float*)take((size_t)128 * 256 * 4);
  float*    cnn2   = (float*)take((size_t)128 * 128 * 4);
  _Float16* xbuf0  = (_Float16*)take((size_t)128 * 8 * 64 * 2);
  _Float16* xbuf1  = (_Float16*)take((size_t)128 * 8 * 64 * 2);

  conv1_kernel<<<C1OUT / 256, 256, 0, stream>>>(image, conv1_w, conv1_b, c1o);

  // conv2: 22816 M-tiles, 37,888B dynamic LDS (full filter + A tile)
  conv_wmma_kernel<32, 94, 126, 46, 62, 1>
      <<<2852, 128, (9 * 64 * 32 + 16 * 32) * 2, stream>>>(
          c1o, conv2_w, conv2_b, c2o, C2OUT / (64 * 16));
  // conv3: 5280 M-tiles, 75,776B dynamic LDS
  conv_wmma_kernel<64, 46, 62, 22, 30, 0>
      <<<1320, 128, (9 * 64 * 64 + 16 * 64) * 2, stream>>>(
          c2o, conv3_w, conv3_b, c3o, C3OUT / (64 * 16));

  cvt_f32_to_f16_kernel<<<(256 * C3K) / 1024, 1024, 0, stream>>>(
      fc1_w, fc1w16, 256 * C3K);
  // fc1: double-buffered TDM, K-chunk 64 -> 68KB dynamic LDS
  fc1_kernel<<<8, 512, (2 * 256 * 64 + 2 * 16 * 64) * 2, stream>>>(
      c3o, fc1w16, fc1_b, fc1o);
  head_mlp_kernel<<<1, 256, 0, stream>>>(fc1o, fc2_w, fc2_b, l1_w, l1_b, l2_w, l2_b, cnn2);
  action_kernel<<<(128 * 8 * 64) / 256, 256, 0, stream>>>(
      act_in, a1_w, a1_b, a2_w, a2_b, xbuf0);
  lstm_kernel<<<8, 128, (2 * 256 * 64 + 16 * 64) * 2, stream>>>(
      cnn2, Wih0, Wih, Whh, bih, bhh, xbuf0, xbuf1);
  output_kernel<<<4, 256, 0, stream>>>(xbuf0, o1_w, o1_b, o2_w, o2_b, gt, (float*)d_out);
}
